// FlashMultiHeadAttention_89575837925626
// MI455X (gfx1250) — matrix operand order repaired and checked by reading, not compiled, not run on hardware
//
#include <hip/hip_runtime.h>
#include <hip/hip_bf16.h>
#include <math.h>

typedef __attribute__((ext_vector_type(16))) __bf16 v16bf;
typedef __attribute__((ext_vector_type(8)))  __bf16 v8bf;
typedef __attribute__((ext_vector_type(4)))  __bf16 v4bf;
typedef __attribute__((ext_vector_type(8)))  float  v8f;
typedef __attribute__((ext_vector_type(4)))  float  v4f;

union BF16x16 { v16bf v; v8bf h[2]; __bf16 e[16]; };
union F32x8   { v8f  v; float  e[8];  };

#define WMMA_BF16(a, b, c) \
  __builtin_amdgcn_wmma_f32_16x16x32_bf16(false, (a), false, (b), (short)0, (c), false, false)

// ---------------------------------------------------------------------------
// Tiled WMMA GEMM: Y = X @ W + bias.  X: MxK, W: KxN f32 (row-major).
// Block = 256 threads = 8 waves (4x2), block tile 128x64, wave tile 32x32.
// Double-buffered LDS with register prefetch: loads for tile t+1 issue before
// the WMMAs of tile t; the loadcnt wait lands after the math.
// ---------------------------------------------------------------------------
template<bool XBF, bool OUTF32>
__global__ __launch_bounds__(256) void gemm_wmma(
    const void* __restrict__ Xv, const float* __restrict__ W,
    const float* __restrict__ bias, void* __restrict__ Yv,
    int M, int N, int K)
{
  __shared__ __bf16 As[2][128][40];    // 128x32 tiles, +8 pad
  __shared__ __bf16 BsT[2][64][40];    // 32x64 tiles stored transposed [n][k]

  const int bm = blockIdx.x, bn = blockIdx.y;
  const int tid  = threadIdx.x;
  const int lane = tid & 31, wave = tid >> 5;
  const int wm = wave & 3, wn = wave >> 2;     // 4 x 2 wave grid
  const int lr = lane & 15, hi = lane >> 4;

  F32x8 acc[2][2];
#pragma unroll
  for (int a = 0; a < 2; ++a)
#pragma unroll
    for (int b = 0; b < 2; ++b)
#pragma unroll
      for (int i = 0; i < 8; ++i) acc[a][b].e[i] = 0.f;

  // fixed staging assignments
  const int am  = tid >> 1;              // A row 0..127
  const int ak0 = (tid & 1) * 16;        // A k-base 0 or 16
  const int bk  = tid >> 3;              // B k-row 0..31
  const int bn8 = (tid & 7) * 8;         // B n-base 0..56

  v4f  arf[4];                           // A prefetch regs (f32 path)
  v8bf arb[2];                           //                 (bf16 path)
  v4f  wrf[2];                           // B prefetch regs

  auto gload = [&](int k0) {
    size_t gi = (size_t)(bm * 128 + am) * K + k0 + ak0;
    if (XBF) {
      const v8bf* p = (const v8bf*)((const __bf16*)Xv + gi);
      arb[0] = p[0]; arb[1] = p[1];
    } else {
      const v4f* p = (const v4f*)((const float*)Xv + gi);
      arf[0] = p[0]; arf[1] = p[1]; arf[2] = p[2]; arf[3] = p[3];
    }
    const v4f* q = (const v4f*)&W[(size_t)(k0 + bk) * N + bn * 64 + bn8];
    wrf[0] = q[0]; wrf[1] = q[1];
  };
  auto stage = [&](int buf) {
    __bf16 tmp[16];
    if (XBF) {
      *(v8bf*)&tmp[0] = arb[0];
      *(v8bf*)&tmp[8] = arb[1];
    } else {
#pragma unroll
      for (int j = 0; j < 4; ++j) {
        tmp[j]      = (__bf16)arf[0][j];
        tmp[4 + j]  = (__bf16)arf[1][j];
        tmp[8 + j]  = (__bf16)arf[2][j];
        tmp[12 + j] = (__bf16)arf[3][j];
      }
    }
    *(v8bf*)&As[buf][am][ak0]     = *(v8bf*)&tmp[0];
    *(v8bf*)&As[buf][am][ak0 + 8] = *(v8bf*)&tmp[8];
#pragma unroll
    for (int j = 0; j < 4; ++j) {
      BsT[buf][bn8 + j][bk]     = (__bf16)wrf[0][j];
      BsT[buf][bn8 + 4 + j][bk] = (__bf16)wrf[1][j];
    }
  };

  gload(0);
  stage(0);

  int buf = 0;
  for (int k0 = 0; k0 < K; k0 += 32, buf ^= 1) {
    const bool more = (k0 + 32) < K;
    if (more) gload(k0 + 32);            // issue next tile's loads first
    __syncthreads();                     // tiles[buf] ready

    BF16x16 afrag[2];
#pragma unroll
    for (int tm = 0; tm < 2; ++tm) {
      const __bf16* p = &As[buf][wm * 32 + tm * 16 + lr][hi * 8];
      afrag[tm].h[0] = *(const v8bf*)p;          // K 0..7  / 8..15
      afrag[tm].h[1] = *(const v8bf*)(p + 16);   // K 16..23/ 24..31
    }
#pragma unroll
    for (int tn = 0; tn < 2; ++tn) {
      const __bf16* q = &BsT[buf][wn * 32 + tn * 16 + lr][hi * 8];
      BF16x16 bfrag;
      bfrag.h[0] = *(const v8bf*)q;
      bfrag.h[1] = *(const v8bf*)(q + 16);
#pragma unroll
      for (int tm = 0; tm < 2; ++tm)
        acc[tm][tn].v = WMMA_BF16(afrag[tm].v, bfrag.v, acc[tm][tn].v);
    }
    if (more) stage(buf ^ 1);            // convert+store after the math
  }

  // ---- epilogue: C layout row = r + 8*hi, col = lr ----
#pragma unroll
  for (int tm = 0; tm < 2; ++tm)
#pragma unroll
    for (int tn = 0; tn < 2; ++tn) {
      int gn = bn * 64 + wn * 32 + tn * 16 + lr;
      float bvv = bias[gn];
#pragma unroll
      for (int r = 0; r < 8; ++r) {
        int gm = bm * 128 + wm * 32 + tm * 16 + r + hi * 8;
        float val = acc[tm][tn].e[r] + bvv;
        if (OUTF32) ((float*)Yv)[(size_t)gm * N + gn]  = val;
        else        ((__bf16*)Yv)[(size_t)gm * N + gn] = (__bf16)val;
      }
    }
}

// ---------------------------------------------------------------------------
// Fusion: action gate + sigmoid gating of V + RoPE on Q/K, per-head relayout,
// plus precompute of the per-(b,h,key) attention bias.
// One block per (b,l) row; thread t owns dims [4t, 4t+4).
// V is written TRANSPOSED per head: Vh[bh][dh][l]  (feeds P@V B-fragments).
// ---------------------------------------------------------------------------
__global__ __launch_bounds__(256) void fuse_gate_rope(
    const __bf16* __restrict__ Qp, const __bf16* __restrict__ Kp,
    const __bf16* __restrict__ Vp, const __bf16* __restrict__ Up,
    const int* __restrict__ action_ids, const float* __restrict__ action_emb,
    const float* __restrict__ Wap, const float* __restrict__ bap,
    const int* __restrict__ td, const float* __restrict__ td_emb,
    const float* __restrict__ td_gate,
    __bf16* __restrict__ Qh, __bf16* __restrict__ Kh, __bf16* __restrict__ VhT,
    float* __restrict__ Bias)
{
  const int L = 2048, D = 1024;
  const int row = blockIdx.x;        // b*L + l
  const int b = row >> 11, l = row & 2047;
  const int tid = threadIdx.x;
  const int d0 = tid * 4;

  __shared__ float qs[1024];
  __shared__ float kss[1024];
  __shared__ float emb[16];

  if (tid < 16) {
    emb[tid] = action_emb[action_ids[row] * 16 + tid];
    // Bias[bh][key] = sigmoid(td_gate) * td_emb[clamp(td[b,key])][h]
    int tdv = td[row];
    tdv = tdv < 0 ? 0 : (tdv > 127 ? 127 : tdv);
    float sgate = 1.f / (1.f + __expf(-td_gate[0]));
    Bias[((size_t)(b * 16 + tid)) * L + l] = sgate * td_emb[tdv * 16 + tid];
  }
  {
    v4bf qv = *(const v4bf*)&Qp[(size_t)row * D + d0];
    v4bf kv = *(const v4bf*)&Kp[(size_t)row * D + d0];
#pragma unroll
    for (int i = 0; i < 4; ++i) { qs[d0 + i] = (float)qv[i]; kss[d0 + i] = (float)kv[i]; }
  }
  __syncthreads();

  const float LN1E4_64 = logf(10000.f) / 64.f;
  const int h = d0 >> 6;
  v4bf uv = *(const v4bf*)&Up[(size_t)row * D + d0];
  v4bf vv = *(const v4bf*)&Vp[(size_t)row * D + d0];
  __bf16 qo4[4], ko4[4];
#pragma unroll
  for (int i = 0; i < 4; ++i) {
    int d = d0 + i, dh = d & 63;
    float g = bap[d];
#pragma unroll
    for (int j = 0; j < 16; ++j) g += emb[j] * Wap[j * D + d];
    float u = (float)uv[i] + g;
    float vgate = (float)vv[i] * (1.f / (1.f + __expf(-u)));
    float invf = __expf(-(float)(dh & ~1) * LN1E4_64);
    float ang = (float)l * invf;
    float c = __cosf(ang), s = __sinf(ang);
    float rq = (dh < 32) ? -qs[h * 64 + dh + 32] : qs[h * 64 + dh - 32];
    float rk = (dh < 32) ? -kss[h * 64 + dh + 32] : kss[h * 64 + dh - 32];
    qo4[i] = (__bf16)(qs[d] * c + rq * s);
    ko4[i] = (__bf16)(kss[d] * c + rk * s);
    VhT[((size_t)(b * 16 + h) * 64 + dh) * L + l] = (__bf16)vgate;   // [bh][dh][l]
  }
  size_t oidx = (((size_t)(b * 16 + h)) * L + l) * 64 + (d0 & 63);
  *(v4bf*)&Qh[oidx] = *(v4bf*)qo4;
  *(v4bf*)&Kh[oidx] = *(v4bf*)ko4;
}

// ---------------------------------------------------------------------------
// Flash attention: one (b,h) head x 64 query rows per block, 4 waves.
// Each wave owns 16 query rows. Online softmax; QK^T and P@V via WMMA.
// K/V tiles double-buffered in LDS with register prefetch.
// ---------------------------------------------------------------------------
__global__ __launch_bounds__(128) void flash_attn_wmma(
    const __bf16* __restrict__ Qh, const __bf16* __restrict__ Kh,
    const __bf16* __restrict__ VhT, const unsigned char* __restrict__ mask,
    const float* __restrict__ Bias, __bf16* __restrict__ AO)
{
  const int L = 2048;
  const int bh = blockIdx.x;            // b*16 + h
  const int b = bh >> 4;
  const int h = bh & 15;
  const int q0 = blockIdx.y * 64;
  const int tid = threadIdx.x;
  const int lane = tid & 31, wave = tid >> 5;
  const int lr = lane & 15, hi = lane >> 4;

  __shared__ __bf16 Qs[64][72];            // [qrow][dh]  (+8 pad)
  __shared__ __bf16 Ks[2][64][72];         // [key][dh]
  __shared__ __bf16 Vt[2][64][72];         // [dh][key]   (V transposed)
  __shared__ __bf16 Ps[4][16][72];         // per-wave P tile [qrow][key]

  const size_t head_base = (size_t)bh * L * 64;
  const int srow = tid >> 1;               // 0..63
  const int shalf = (tid & 1) * 32;        // 0 or 32

  // ---- stage Q tile (32 contiguous bf16 per thread = 4 x b128) ----
  {
    const v8bf* p = (const v8bf*)&Qh[head_base + (size_t)(q0 + srow) * 64 + shalf];
#pragma unroll
    for (int i = 0; i < 4; ++i) *(v8bf*)&Qs[srow][shalf + i * 8] = p[i];
  }

  v8bf krg[4], vrg[4];
  auto gloadKV = [&](int kb) {
    const v8bf* pk = (const v8bf*)&Kh[head_base + (size_t)(kb + srow) * 64 + shalf];
    const v8bf* pv = (const v8bf*)&VhT[head_base + (size_t)srow * L + kb + shalf];
#pragma unroll
    for (int i = 0; i < 4; ++i) { krg[i] = pk[i]; vrg[i] = pv[i]; }
  };
  auto stageKV = [&](int buf) {
#pragma unroll
    for (int i = 0; i < 4; ++i) {
      *(v8bf*)&Ks[buf][srow][shalf + i * 8] = krg[i];
      *(v8bf*)&Vt[buf][srow][shalf + i * 8] = vrg[i];
    }
  };

  gloadKV(0);
  stageKV(0);
  __syncthreads();

  // ---- Q A-fragments (2 k-steps over DH=64), contiguous b128 chunks ----
  BF16x16 qa[2];
#pragma unroll
  for (int ks = 0; ks < 2; ++ks) {
    const __bf16* p = &Qs[wave * 16 + lr][ks * 32 + hi * 8];
    qa[ks].h[0] = *(const v8bf*)p;
    qa[ks].h[1] = *(const v8bf*)(p + 16);
  }

  float mrow[8], lsum[8];
  F32x8 oacc[4];
#pragma unroll
  for (int r = 0; r < 8; ++r) { mrow[r] = -3.0e38f; lsum[r] = 0.f; }
#pragma unroll
  for (int t = 0; t < 4; ++t)
#pragma unroll
    for (int r = 0; r < 8; ++r) oacc[t].e[r] = 0.f;

  const float scale = 0.125f;   // 64^-0.5
  const float* biasrow = &Bias[(size_t)bh * L];

  int buf = 0;
  for (int kb = 0; kb < L; kb += 64, buf ^= 1) {
    const bool more = (kb + 64) < L;
    if (more) gloadKV(kb + 64);          // issue next tile's loads first
    if (kb) __syncthreads();             // tiles[buf] ready

    // ---- S = Q K^T (4 tiles of 16 keys) + scale + bias + mask ----
    float pfrag[4][8];
#pragma unroll
    for (int ct = 0; ct < 4; ++ct) {
      F32x8 s;
#pragma unroll
      for (int r = 0; r < 8; ++r) s.e[r] = 0.f;
#pragma unroll
      for (int ks = 0; ks < 2; ++ks) {
        const __bf16* p = &Ks[buf][ct * 16 + lr][ks * 32 + hi * 8];
        BF16x16 kf;
        kf.h[0] = *(const v8bf*)p;
        kf.h[1] = *(const v8bf*)(p + 16);
        s.v = WMMA_BF16(qa[ks].v, kf.v, s.v);
      }
      int key = kb + ct * 16 + lr;
      float biasv = biasrow[key];
      const unsigned char* mrow_p =
          &mask[((size_t)b * L + q0 + wave * 16 + hi * 8) * L + key];
#pragma unroll
      for (int r = 0; r < 8; ++r) {
        float sv = s.e[r] * scale + biasv;
        if (!mrow_p[(size_t)r * L]) sv = -3.0e38f;
        pfrag[ct][r] = sv;
      }
    }

    // ---- online softmax over this 64-key block ----
    float mnew[8], alpha[8];
#pragma unroll
    for (int r = 0; r < 8; ++r) {
      float mx = fmaxf(fmaxf(pfrag[0][r], pfrag[1][r]),
                       fmaxf(pfrag[2][r], pfrag[3][r]));
#pragma unroll
      for (int d = 1; d < 16; d <<= 1) mx = fmaxf(mx, __shfl_xor(mx, d, 32));
      mnew[r] = fmaxf(mrow[r], mx);
      alpha[r] = __expf(mrow[r] - mnew[r]);
      mrow[r] = mnew[r];
    }
    float rsum[8];
#pragma unroll
    for (int r = 0; r < 8; ++r) rsum[r] = 0.f;
#pragma unroll
    for (int ct = 0; ct < 4; ++ct)
#pragma unroll
      for (int r = 0; r < 8; ++r) {
        float p = __expf(pfrag[ct][r] - mnew[r]);
        pfrag[ct][r] = p;
        rsum[r] += p;
      }
#pragma unroll
    for (int r = 0; r < 8; ++r) {
      float sm = rsum[r];
#pragma unroll
      for (int d = 1; d < 16; d <<= 1) sm += __shfl_xor(sm, d, 32);
      lsum[r] = lsum[r] * alpha[r] + sm;
    }
#pragma unroll
    for (int t = 0; t < 4; ++t)
#pragma unroll
      for (int r = 0; r < 8; ++r) oacc[t].e[r] *= alpha[r];

    // ---- relayout P (C-frag -> A-frag) via LDS ----
#pragma unroll
    for (int ct = 0; ct < 4; ++ct)
#pragma unroll
      for (int r = 0; r < 8; ++r)
        Ps[wave][r + hi * 8][ct * 16 + lr] = (__bf16)pfrag[ct][r];
    __syncthreads();

    BF16x16 pa[2];
#pragma unroll
    for (int ks = 0; ks < 2; ++ks) {
      const __bf16* p = &Ps[wave][lr][ks * 32 + hi * 8];
      pa[ks].h[0] = *(const v8bf*)p;
      pa[ks].h[1] = *(const v8bf*)(p + 16);
    }

    // ---- O += P @ V  (V B-fragments contiguous from transposed tile) ----
#pragma unroll
    for (int dt = 0; dt < 4; ++dt) {
#pragma unroll
      for (int ks = 0; ks < 2; ++ks) {
        const __bf16* p = &Vt[buf][dt * 16 + lr][ks * 32 + hi * 8];
        BF16x16 vf;
        vf.h[0] = *(const v8bf*)p;
        vf.h[1] = *(const v8bf*)(p + 16);
        oacc[dt].v = WMMA_BF16(pa[ks].v, vf.v, oacc[dt].v);
      }
    }
    if (more) stageKV(buf ^ 1);          // store next tiles after the math
  }

  // ---- normalize & store to [B, L, H*DH] bf16 ----
#pragma unroll
  for (int dt = 0; dt < 4; ++dt)
#pragma unroll
    for (int r = 0; r < 8; ++r) {
      int qq = q0 + wave * 16 + r + hi * 8;
      int dh = dt * 16 + lr;
      float val = oacc[dt].e[r] / lsum[r];
      AO[((size_t)(b * L + qq)) * 1024 + h * 64 + dh] = (__bf16)val;
    }
}

// ---------------------------------------------------------------------------
extern "C" void kernel_launch(void* const* d_in, const int* in_sizes, int n_in,
                              void* d_out, int out_size, void* d_ws, size_t ws_size,
                              hipStream_t stream)
{
  const float* query = (const float*)d_in[0];
  const float* key   = (const float*)d_in[1];
  const float* value = (const float*)d_in[2];
  const unsigned char* attn_mask = (const unsigned char*)d_in[3];
  const int* action_ids  = (const int*)d_in[4];
  const int* time_deltas = (const int*)d_in[5];
  const float* Wq = (const float*)d_in[6];  const float* bq = (const float*)d_in[7];
  const float* Wk = (const float*)d_in[8];  const float* bk = (const float*)d_in[9];
  const float* Wv = (const float*)d_in[10]; const float* bv = (const float*)d_in[11];
  const float* Wu = (const float*)d_in[12]; const float* bu = (const float*)d_in[13];
  const float* Wo = (const float*)d_in[14]; const float* bo = (const float*)d_in[15];
  const float* action_emb = (const float*)d_in[16];
  const float* Wap = (const float*)d_in[17];
  const float* bap = (const float*)d_in[18];
  const float* td_emb  = (const float*)d_in[19];
  const float* td_gate = (const float*)d_in[20];

  const int B = 2, L = 2048, D = 1024, M = B * L;
  char* ws = (char*)d_ws;
  const size_t SZ = (size_t)M * D * sizeof(__bf16);   // 8 MB each
  __bf16* Qp  = (__bf16*)(ws + 0 * SZ);
  __bf16* Kp  = (__bf16*)(ws + 1 * SZ);
  __bf16* Vp  = (__bf16*)(ws + 2 * SZ);
  __bf16* Up  = (__bf16*)(ws + 3 * SZ);
  __bf16* Qh  = (__bf16*)(ws + 4 * SZ);
  __bf16* Kh  = (__bf16*)(ws + 5 * SZ);
  __bf16* VhT = (__bf16*)(ws + 6 * SZ);
  __bf16* AO  = (__bf16*)(ws + 7 * SZ);
  float*  Bias = (float*)(ws + 8 * SZ);               // 32*2048 f32 = 256 KB

  dim3 ggrid(M / 128, D / 64), gblk(256);
  gemm_wmma<false, false><<<ggrid, gblk, 0, stream>>>(query, Wq, bq, Qp, M, D, D);
  gemm_wmma<false, false><<<ggrid, gblk, 0, stream>>>(key,   Wk, bk, Kp, M, D, D);
  gemm_wmma<false, false><<<ggrid, gblk, 0, stream>>>(value, Wv, bv, Vp, M, D, D);
  gemm_wmma<false, false><<<ggrid, gblk, 0, stream>>>(query, Wu, bu, Up, M, D, D);

  fuse_gate_rope<<<dim3(M), dim3(256), 0, stream>>>(
      Qp, Kp, Vp, Up, action_ids, action_emb, Wap, bap,
      time_deltas, td_emb, td_gate, Qh, Kh, VhT, Bias);

  flash_attn_wmma<<<dim3(B * 16, L / 64), dim3(128), 0, stream>>>(
      Qh, Kh, VhT, attn_mask, Bias, AO);

  gemm_wmma<true, true><<<ggrid, gblk, 0, stream>>>(AO, Wo, bo, (float*)d_out, M, D, D);
}
